// Seq2SeqModel_44753559225087
// MI455X (gfx1250) — matrix operand order repaired and checked by reading, not compiled, not run on hardware
//
#include <hip/hip_runtime.h>

// Problem constants
#define BB 32
#define SS 64
#define STt 64
#define EE 512
#define HH 1024
#define GG 4096   // 4*H
#define VV 32000

typedef __bf16 v16bf __attribute__((ext_vector_type(16)));
typedef float  v8f   __attribute__((ext_vector_type(8)));
typedef unsigned short u16x16 __attribute__((ext_vector_type(16)));

__device__ __forceinline__ float bf2f(unsigned short h) {
    unsigned int u = ((unsigned int)h) << 16;
    return __builtin_bit_cast(float, u);
}
__device__ __forceinline__ unsigned short f2bf(float f) {
    unsigned int u = __builtin_bit_cast(unsigned int, f);
    u += 0x7fffu + ((u >> 16) & 1u);   // round-to-nearest-even
    return (unsigned short)(u >> 16);
}
__device__ __forceinline__ float sigm(float x) {
    return 1.0f / (1.0f + __expf(-x));
}

// ---------------- fp32 -> bf16 weight conversion ----------------
__global__ void f32_to_bf16_kernel(const float* __restrict__ in,
                                   unsigned short* __restrict__ out, long long n) {
    long long i = (long long)blockIdx.x * blockDim.x + threadIdx.x;
    if (i < n) out[i] = f2bf(in[i]);
}

// ---------------- embedding gather (int idx -> bf16 rows) ----------------
__global__ void gather_embed_kernel(const int* __restrict__ idx,
                                    const float* __restrict__ emb,
                                    unsigned short* __restrict__ out, long long total) {
    long long i = (long long)blockIdx.x * blockDim.x + threadIdx.x;
    if (i >= total) return;
    int e = (int)(i % EE);
    long long r = i / EE;
    int row = idx[r];
    out[i] = f2bf(emb[(long long)row * EE + e]);
}

// ---------------- bf16 WMMA GEMM: out = A[M,K] @ W[N,K]^T (+bias)(+addsrc) ----------
// One wave computes a 16x64 tile (4 accumulators), K in steps of 32.
// A, W row-major bf16. Optional fp32 bias[N], fp32 addsrc[M,N] (accumulator init).
// Outputs: fp32 (ldf) and/or bf16 (ldh). All branches wave-uniform -> EXEC all ones.
__global__ void gemm_wmma_bf16_kernel(
    const unsigned short* __restrict__ A, int lda,
    const unsigned short* __restrict__ Bw, int ldb,
    const float* __restrict__ bias,
    const float* __restrict__ addsrc, long long ld_add,
    float* __restrict__ outF, long long ldf,
    unsigned short* __restrict__ outH, long long ldh,
    int M, int N, int K)
{
    int wave = (int)((blockIdx.x * blockDim.x + threadIdx.x) >> 5);
    int lane = threadIdx.x & 31;
    int tilesN = N >> 6;
    int tilesM = M >> 4;
    if (wave >= tilesM * tilesN) return;
    int tm = wave / tilesN, tn = wave % tilesN;
    int row0 = tm << 4, col0 = tn << 6;
    int nlane = lane & 15;
    int mhalf = (lane < 16) ? 0 : 8;

    v8f acc[4];
    for (int j = 0; j < 4; ++j) {
        int n = col0 + j * 16 + nlane;
        float binit = bias ? bias[n] : 0.0f;
        for (int r = 0; r < 8; ++r) {
            float v = binit;
            if (addsrc) v += addsrc[(long long)(row0 + mhalf + r) * ld_add + n];
            acc[j][r] = v;
        }
    }

    // A fragment K-offsets per 16-bit A 16x32 layout:
    //   lanes 0-15: halves 0..7 -> K 0..7, halves 8..15 -> K 16..23
    //   lanes16-31: halves 0..7 -> K 8..15, halves 8..15 -> K 24..31
    int kbA = (lane < 16) ? 0 : 8;
    int kbB = (lane < 16) ? 0 : 8;    // B 32x16: same K order as A

    for (int k0 = 0; k0 < K; k0 += 32) {
        const unsigned short* pa = A + (long long)(row0 + nlane) * lda + k0;
        u16x16 au;
        #pragma unroll
        for (int q = 0; q < 8; ++q) au[q] = pa[kbA + q];
        #pragma unroll
        for (int q = 0; q < 8; ++q) au[8 + q] = pa[kbA + 16 + q];
        v16bf a = __builtin_bit_cast(v16bf, au);

        #pragma unroll
        for (int j = 0; j < 4; ++j) {
            const unsigned short* pb =
                Bw + (long long)(col0 + j * 16 + nlane) * ldb + k0 + kbB;
            u16x16 bu;
            #pragma unroll
            for (int q = 0; q < 8; ++q) { bu[q] = pb[q]; bu[8 + q] = pb[16 + q]; }
            v16bf b = __builtin_bit_cast(v16bf, bu);
            acc[j] = __builtin_amdgcn_wmma_f32_16x16x32_bf16(
                false, a, false, b, (short)0, acc[j], false, false);
        }
    }

    for (int j = 0; j < 4; ++j) {
        int n = col0 + j * 16 + nlane;
        for (int r = 0; r < 8; ++r) {
            long long row = row0 + mhalf + r;
            float v = acc[j][r];
            if (outF) outF[row * ldf + n] = v;
            if (outH) outH[row * ldh + n] = f2bf(v);
        }
    }
}

// ---------------- LSTM gate nonlinearity + state update ----------------
// gates fp32 [B, 4H] (i,f,g,o). c fp32 in/out, h bf16 out; optional 2nd h copy.
__global__ void lstm_elem_kernel(const float* __restrict__ gates,
                                 float* __restrict__ c,
                                 unsigned short* __restrict__ h,
                                 unsigned short* __restrict__ hout2,
                                 long long h2_rowstride) {
    int idx = blockIdx.x * blockDim.x + threadIdx.x;   // B*H = 32768
    int b = idx >> 10, j = idx & 1023;
    const float* g = gates + (long long)b * GG;
    float i_ = sigm(g[j]);
    float f_ = sigm(g[HH + j]);
    float gg = tanhf(g[2 * HH + j]);
    float o_ = sigm(g[3 * HH + j]);
    float cn = f_ * c[idx] + i_ * gg;
    c[idx] = cn;
    unsigned short hn = f2bf(o_ * tanhf(cn));
    h[idx] = hn;
    if (hout2) hout2[(long long)b * h2_rowstride + j] = hn;
}

// ---------------- fused attention: scores -> softmax -> context ----------------
// One block per batch element. q = decoder-L1 h (bf16), keys/values bf16.
__global__ void attn_kernel(const unsigned short* __restrict__ q,       // [B,H]
                            const unsigned short* __restrict__ key,     // [B,Ss,H]
                            const unsigned short* __restrict__ encOut,  // [B,Ss,H]
                            unsigned short* __restrict__ ctx) {         // [B,H]
    int b = blockIdx.x;
    int tid = threadIdx.x;                 // 256 threads
    __shared__ float sc[SS];
    __shared__ float part[256];

    int s = tid >> 2, p = tid & 3;         // 4 threads per source position
    const unsigned short* qp = q + (long long)b * HH;
    const unsigned short* kp = key + ((long long)b * SS + s) * HH;
    float sum = 0.0f;
    for (int h = p * 256; h < p * 256 + 256; ++h)
        sum += bf2f(qp[h]) * bf2f(kp[h]);
    part[tid] = sum;
    __syncthreads();
    if (p == 0) sc[s] = part[tid] + part[tid + 1] + part[tid + 2] + part[tid + 3];
    __syncthreads();
    if (tid == 0) {
        float m = sc[0];
        for (int i = 1; i < SS; ++i) m = fmaxf(m, sc[i]);
        float tot = 0.0f;
        for (int i = 0; i < SS; ++i) { float e = __expf(sc[i] - m); sc[i] = e; tot += e; }
        float inv = 1.0f / tot;
        for (int i = 0; i < SS; ++i) sc[i] *= inv;
    }
    __syncthreads();
    for (int h = tid; h < HH; h += 256) {
        float acc = 0.0f;
        const unsigned short* vp = encOut + (long long)b * SS * HH + h;
        for (int si = 0; si < SS; ++si)
            acc += sc[si] * bf2f(vp[(long long)si * HH]);
        ctx[(long long)b * HH + h] = f2bf(acc);
    }
}

// =======================================================================
extern "C" void kernel_launch(void* const* d_in, const int* in_sizes, int n_in,
                              void* d_out, int out_size, void* d_ws, size_t ws_size,
                              hipStream_t stream) {
    (void)in_sizes; (void)n_in; (void)out_size; (void)ws_size;
    const int*   source   = (const int*)d_in[0];
    const int*   prev     = (const int*)d_in[1];
    const float* emb      = (const float*)d_in[2];
    const float* encWih0f = (const float*)d_in[3];
    const float* encWhh0f = (const float*)d_in[4];
    const float* enc_b0   = (const float*)d_in[5];
    const float* encWih1f = (const float*)d_in[6];
    const float* encWhh1f = (const float*)d_in[7];
    const float* enc_b1   = (const float*)d_in[8];
    const float* decWih0f = (const float*)d_in[9];
    const float* decWhh0f = (const float*)d_in[10];
    const float* dec_b0   = (const float*)d_in[11];
    const float* decWih1f = (const float*)d_in[12];
    const float* decWhh1f = (const float*)d_in[13];
    const float* dec_b1   = (const float*)d_in[14];
    const float* Wkf      = (const float*)d_in[15];
    const float* bk       = (const float*)d_in[16];
    const float* Whf      = (const float*)d_in[17];
    const float* bh       = (const float*)d_in[18];
    float* out = (float*)d_out;

    // ----- workspace carving -----
    char* base = (char*)d_ws;
    size_t off = 0;
    auto carve = [&](size_t bytes) -> void* {
        void* p = base + off;
        off = (off + bytes + 255) & ~(size_t)255;
        return p;
    };
    unsigned short* wEncWih0 = (unsigned short*)carve((size_t)GG * EE * 2);
    unsigned short* wEncWhh0 = (unsigned short*)carve((size_t)GG * HH * 2);
    unsigned short* wEncWih1 = (unsigned short*)carve((size_t)GG * HH * 2);
    unsigned short* wEncWhh1 = (unsigned short*)carve((size_t)GG * HH * 2);
    unsigned short* wDecWih0 = (unsigned short*)carve((size_t)GG * (EE + HH) * 2);
    unsigned short* wDecWhh0 = (unsigned short*)carve((size_t)GG * HH * 2);
    unsigned short* wDecWih1 = (unsigned short*)carve((size_t)GG * HH * 2);
    unsigned short* wDecWhh1 = (unsigned short*)carve((size_t)GG * HH * 2);
    unsigned short* wWk      = (unsigned short*)carve((size_t)HH * HH * 2);
    unsigned short* wWh      = (unsigned short*)carve((size_t)VV * HH * 2);
    unsigned short* srcE     = (unsigned short*)carve((size_t)BB * SS * EE * 2);
    unsigned short* tgtE     = (unsigned short*)carve((size_t)BB * STt * EE * 2);
    float*          xgE      = (float*)carve((size_t)BB * SS * GG * 4);
    float*          xgD      = (float*)carve((size_t)BB * STt * GG * 4);
    unsigned short* encOut   = (unsigned short*)carve((size_t)BB * SS * HH * 2);
    unsigned short* attnKey  = (unsigned short*)carve((size_t)BB * SS * HH * 2);
    unsigned short* h0       = (unsigned short*)carve((size_t)BB * HH * 2);
    unsigned short* h1       = (unsigned short*)carve((size_t)BB * HH * 2);
    float*          c0       = (float*)carve((size_t)BB * HH * 4);
    float*          c1       = (float*)carve((size_t)BB * HH * 4);
    float*          gates    = (float*)carve((size_t)BB * GG * 4);
    unsigned short* ctx      = (unsigned short*)carve((size_t)BB * HH * 2);

    auto conv = [&](const float* s, unsigned short* d, long long n) {
        int blocks = (int)((n + 255) / 256);
        f32_to_bf16_kernel<<<blocks, 256, 0, stream>>>(s, d, n);
    };
    auto gemm = [&](const unsigned short* A, int lda,
                    const unsigned short* Bw, int ldb,
                    const float* bias, const float* addsrc, long long ld_add,
                    float* oF, long long ldf, unsigned short* oH, long long ldh,
                    int M, int N, int K) {
        int waves = (M >> 4) * (N >> 6);
        int blocks = (waves * 32 + 255) / 256;
        gemm_wmma_bf16_kernel<<<blocks, 256, 0, stream>>>(
            A, lda, Bw, ldb, bias, addsrc, ld_add, oF, ldf, oH, ldh, M, N, K);
    };

    // ----- weight/bf16 prep (keeps hot weights at half size -> resident in 192MB L2) -----
    conv(encWih0f, wEncWih0, (long long)GG * EE);
    conv(encWhh0f, wEncWhh0, (long long)GG * HH);
    conv(encWih1f, wEncWih1, (long long)GG * HH);
    conv(encWhh1f, wEncWhh1, (long long)GG * HH);
    conv(decWih0f, wDecWih0, (long long)GG * (EE + HH));
    conv(decWhh0f, wDecWhh0, (long long)GG * HH);
    conv(decWih1f, wDecWih1, (long long)GG * HH);
    conv(decWhh1f, wDecWhh1, (long long)GG * HH);
    conv(Wkf, wWk, (long long)HH * HH);
    conv(Whf, wWh, (long long)VV * HH);

    {
        long long tot = (long long)BB * SS * EE;
        gather_embed_kernel<<<(int)((tot + 255) / 256), 256, 0, stream>>>(source, emb, srcE, tot);
        gather_embed_kernel<<<(int)((tot + 255) / 256), 256, 0, stream>>>(prev, emb, tgtE, tot);
    }

    // zero initial states
    hipMemsetAsync(h0, 0, (size_t)BB * HH * 2, stream);
    hipMemsetAsync(h1, 0, (size_t)BB * HH * 2, stream);
    hipMemsetAsync(c0, 0, (size_t)BB * HH * 4, stream);
    hipMemsetAsync(c1, 0, (size_t)BB * HH * 4, stream);

    // ----- non-recurrent big GEMMs (all timesteps at once) -----
    // xgE[b*Ss+s, :] = src_e @ encWih0^T + enc_b0
    gemm(srcE, EE, wEncWih0, EE, enc_b0, nullptr, 0,
         xgE, GG, nullptr, 0, BB * SS, GG, EE);
    // xgD = tgt_e @ decWih0[:, :E]^T + dec_b0   (column split of concat weight)
    gemm(tgtE, EE, wDecWih0, EE + HH, dec_b0, nullptr, 0,
         xgD, GG, nullptr, 0, BB * STt, GG, EE);

    // ----- encoder recurrence -----
    for (int t = 0; t < SS; ++t) {
        // L0: gates = xgE[:,t,:] + h0 @ Whh0^T
        gemm(h0, HH, wEncWhh0, HH, nullptr, xgE + (long long)t * GG, (long long)SS * GG,
             gates, GG, nullptr, 0, BB, GG, HH);
        lstm_elem_kernel<<<128, 256, 0, stream>>>(gates, c0, h0, nullptr, 0);
        // L1: gates = h0 @ Wih1^T + b1 ; gates += h1 @ Whh1^T
        gemm(h0, HH, wEncWih1, HH, enc_b1, nullptr, 0,
             gates, GG, nullptr, 0, BB, GG, HH);
        gemm(h1, HH, wEncWhh1, HH, nullptr, gates, GG,
             gates, GG, nullptr, 0, BB, GG, HH);
        lstm_elem_kernel<<<128, 256, 0, stream>>>(gates, c1, h1,
            encOut + (long long)t * HH, (long long)SS * HH);
    }

    // attn_key = enc_out @ Wk^T + bk  (bf16 out)
    gemm(encOut, HH, wWk, HH, bk, nullptr, 0,
         nullptr, 0, attnKey, HH, BB * SS, HH, HH);

    // ----- decoder recurrence (states carry over from encoder finals) -----
    for (int t = 0; t < STt; ++t) {
        attn_kernel<<<BB, 256, 0, stream>>>(h1, attnKey, encOut, ctx);
        // L0: gates = xgD[:,t,:] + ctx @ Wih0[:,E:]^T ; gates += h0 @ Whh0^T
        gemm(ctx, HH, wDecWih0 + EE, EE + HH, nullptr,
             xgD + (long long)t * GG, (long long)STt * GG,
             gates, GG, nullptr, 0, BB, GG, HH);
        gemm(h0, HH, wDecWhh0, HH, nullptr, gates, GG,
             gates, GG, nullptr, 0, BB, GG, HH);
        lstm_elem_kernel<<<128, 256, 0, stream>>>(gates, c0, h0, nullptr, 0);
        // L1
        gemm(h0, HH, wDecWih1, HH, dec_b1, nullptr, 0,
             gates, GG, nullptr, 0, BB, GG, HH);
        gemm(h1, HH, wDecWhh1, HH, nullptr, gates, GG,
             gates, GG, nullptr, 0, BB, GG, HH);
        lstm_elem_kernel<<<128, 256, 0, stream>>>(gates, c1, h1, nullptr, 0);
        // logits_t = h1 @ Wh^T + bh -> out[:, t, :]
        gemm(h1, HH, wWh, HH, bh, nullptr, 0,
             out + (long long)t * VV, (long long)STt * VV, nullptr, 0,
             BB, VV, HH);
    }
}
